// TernaryCreditRiskNet_82008105549839
// MI455X (gfx1250) — matrix operand order repaired and checked by reading, not compiled, not run on hardware
//
#include <hip/hip_runtime.h>
#include <cstdint>
#include <cstddef>

// ---------------------------------------------------------------------------
// TernaryCreditRiskNet for MI455X (gfx1250, wave32, WMMA)
//   gather/pool (HBM-bound) -> async-LDS double-buffered bf16 WMMA GEMMs
//   -> LN+GELU -> logits
// ---------------------------------------------------------------------------

#define VOCAB  50257
#define EMBED  1024
#define HIDDEN 4096
#define HALFH  2048
#define NCLS   3
#define BATCH  4096
#define SEQ    128

typedef __attribute__((ext_vector_type(16))) __bf16 v16bf;
typedef __attribute__((ext_vector_type(8)))  float  v8f;

__device__ __forceinline__ unsigned short f2bf(float f) {
    unsigned u = __float_as_uint(f);
    u += 0x7FFFu + ((u >> 16) & 1u);      // round-to-nearest-even
    return (unsigned short)(u >> 16);
}
__device__ __forceinline__ float bf2f(unsigned short h) {
    return __uint_as_float(((unsigned)h) << 16);
}

// Async global->LDS 16B copy (CDNA5 GLOBAL_LOAD_ASYNC_TO_LDS_B128, ASYNCcnt).
// lds_off: raw LDS byte offset (low 32 bits of the generic shared pointer —
// the shared aperture is 4GB-aligned, ISA §10.2). gaddr: 64-bit global addr.
__device__ __forceinline__ void async_copy16(unsigned lds_off, const void* gptr) {
    asm volatile("global_load_async_to_lds_b128 %0, %1, off"
                 :: "v"(lds_off), "v"((unsigned long long)(uintptr_t)gptr)
                 : "memory");
}

// ---------------- mean(|w|) : deterministic two-pass reduction --------------
__global__ __launch_bounds__(256)
void k_abs_partial(const float* __restrict__ w, int n, float* __restrict__ part) {
    __shared__ float red[256];
    int base = blockIdx.x * 4096 + threadIdx.x;
    float s = 0.f;
#pragma unroll
    for (int i = 0; i < 16; ++i) {
        int idx = base + i * 256;
        if (idx < n) s += fabsf(w[idx]);
    }
    red[threadIdx.x] = s;
    __syncthreads();
    for (int off = 128; off > 0; off >>= 1) {
        if ((int)threadIdx.x < off) red[threadIdx.x] += red[threadIdx.x + off];
        __syncthreads();
    }
    if (threadIdx.x == 0) part[blockIdx.x] = red[0];
}

__global__ __launch_bounds__(256)
void k_finalize_scales(const float* __restrict__ p1, int nb1, float inv1,
                       const float* __restrict__ p2, int nb2, float inv2,
                       const float* __restrict__ p3, int nb3, float inv3,
                       float* __restrict__ scales) {
    __shared__ float red[256];
    const float* ps[3] = {p1, p2, p3};
    int   nbs[3]  = {nb1, nb2, nb3};
    float invs[3] = {inv1, inv2, inv3};
    for (int j = 0; j < 3; ++j) {
        float s = 0.f;
        for (int i = threadIdx.x; i < nbs[j]; i += 256) s += ps[j][i];
        red[threadIdx.x] = s;
        __syncthreads();
        for (int off = 128; off > 0; off >>= 1) {
            if ((int)threadIdx.x < off) red[threadIdx.x] += red[threadIdx.x + off];
            __syncthreads();
        }
        if (threadIdx.x == 0) scales[j] = red[0] * invs[j];
        __syncthreads();
    }
}

// ------------- ternary quantize: round(clip(w/s,-1,1)) stored as bf16 -------
__global__ __launch_bounds__(256)
void k_ternary(const float* __restrict__ w, int n, const float* __restrict__ scale,
               unsigned short* __restrict__ out) {
    float inv_s = 1.0f / *scale;
    int base = blockIdx.x * 1024 + threadIdx.x * 4;
#pragma unroll
    for (int i = 0; i < 4; ++i) {
        int idx = base + i;
        if (idx < n) {
            float q = rintf(fminf(fmaxf(w[idx] * inv_s, -1.f), 1.f));   // {-1,0,1}
            out[idx] = f2bf(q);                                         // exact in bf16
        }
    }
}

// -------------------- embedding gather + masked mean pool -------------------
__global__ __launch_bounds__(256)
void k_pool(const int* __restrict__ ids, const float* __restrict__ emb,
            unsigned short* __restrict__ X) {
    __shared__ int sid[SEQ];
    int b = blockIdx.x;
    if (threadIdx.x < SEQ) sid[threadIdx.x] = ids[(size_t)b * SEQ + threadIdx.x];
    __syncthreads();
    float4 acc = {0.f, 0.f, 0.f, 0.f};
    int cnt = 0;
    int col = threadIdx.x * 4;
    for (int s = 0; s < SEQ; ++s) {
        int id = sid[s];
        if (id != 0) {
            const float4 e = *(const float4*)&emb[(size_t)id * EMBED + col];
            acc.x += e.x; acc.y += e.y; acc.z += e.z; acc.w += e.w;
            ++cnt;
        }
    }
    float inv = 1.0f / (float)(cnt < 1 ? 1 : cnt);
    size_t o = (size_t)b * EMBED + col;
    X[o + 0] = f2bf(acc.x * inv);
    X[o + 1] = f2bf(acc.y * inv);
    X[o + 2] = f2bf(acc.z * inv);
    X[o + 3] = f2bf(acc.w * inv);
}

// ---------------- bf16 WMMA GEMM:  C[M,N] = s*(A @ Bw^T) + bias -------------
// A: M x K bf16 row-major.  Bw: N x K bf16 row-major (ternary).  C: f32.
// Block tile 128x128, 8 waves (2x4), wave tile 64x32 -> 4x2 16x16 WMMA accs.
// Double-buffered LDS fed by GLOBAL_LOAD_ASYNC_TO_LDS_B128 (ASYNCcnt).
__global__ __launch_bounds__(256)
void k_gemm_bf16(const unsigned short* __restrict__ A,
                 const unsigned short* __restrict__ Bw,
                 float* __restrict__ C,
                 int M, int N, int K,
                 const float* __restrict__ scale,
                 const float* __restrict__ bias) {
    __shared__ __align__(16) unsigned short As[2][128][40];   // +8 pad rows (80B stride)
    __shared__ __align__(16) unsigned short Bs[2][128][40];

    const int tid  = threadIdx.x;
    const int lane = tid & 31;
    const int wave = tid >> 5;
    const int wm   = (wave & 1) * 64;    // wave M offset in block tile
    const int wn   = (wave >> 1) * 32;   // wave N offset in block tile
    const int m0   = blockIdx.x * 128;
    const int n0   = blockIdx.y * 128;
    const int half = lane >> 4;
    const int l16  = lane & 15;

    // per-thread copy assignment: 2 x 16B chunks per tile
    const int r0 = tid >> 2;                  // chunk tid
    const int c0 = (tid & 3) * 8;
    const int r1 = (tid + 256) >> 2;          // chunk tid+256
    const int c1 = ((tid + 256) & 3) * 8;

    auto issue_tile = [&](int buf, int k0) {
        async_copy16((unsigned)(uintptr_t)&As[buf][r0][c0],
                     &A[(size_t)(m0 + r0) * K + k0 + c0]);
        async_copy16((unsigned)(uintptr_t)&As[buf][r1][c1],
                     &A[(size_t)(m0 + r1) * K + k0 + c1]);
        async_copy16((unsigned)(uintptr_t)&Bs[buf][r0][c0],
                     &Bw[(size_t)(n0 + r0) * K + k0 + c0]);
        async_copy16((unsigned)(uintptr_t)&Bs[buf][r1][c1],
                     &Bw[(size_t)(n0 + r1) * K + k0 + c1]);
    };

    const v8f vzero = {0.f, 0.f, 0.f, 0.f, 0.f, 0.f, 0.f, 0.f};
    v8f acc[4][2];
#pragma unroll
    for (int i = 0; i < 4; ++i)
#pragma unroll
        for (int j = 0; j < 2; ++j) acc[i][j] = vzero;

    const int nk = K >> 5;
    issue_tile(0, 0);                          // prologue: fill buffer 0

    for (int it = 0; it < nk; ++it) {
        const int buf = it & 1;
        if (it + 1 < nk) {
            issue_tile(buf ^ 1, (it + 1) << 5);        // prefetch next tile
            // 8 outstanding; in-order completion => current tile's 4 done at <=4
            asm volatile("s_wait_asynccnt 0x4" ::: "memory");
        } else {
            asm volatile("s_wait_asynccnt 0x0" ::: "memory");
        }
        __syncthreads();                                // all waves' writes visible

        union Fr { v16bf v; unsigned u[8]; };
        Fr fa[4], fb[2];
#pragma unroll
        for (int v = 0; v < 8; ++v) {
            // A layout (ISA 7.12.2, 16-bit A 16x32): lanes 0-15 K={0..7,16..23},
            // lanes 16-31 K={8..15,24..31}, two bf16 per VGPR.
            int ka = ((v & 4) ? 16 : 0) + 8 * half + 2 * (v & 3);
            // B layout (K x N): same K order as A: lanes 0-15 K={0..7,16..23}, lanes 16-31 K={8..15,24..31}.
            int kb = ((v & 4) ? 16 : 0) + 8 * half + 2 * (v & 3);
#pragma unroll
            for (int ms = 0; ms < 4; ++ms)
                fa[ms].u[v] = *(const unsigned*)&As[buf][wm + ms * 16 + l16][ka];
#pragma unroll
            for (int ns = 0; ns < 2; ++ns)
                fb[ns].u[v] = *(const unsigned*)&Bs[buf][wn + ns * 16 + l16][kb];
        }

#pragma unroll
        for (int ms = 0; ms < 4; ++ms)
#pragma unroll
            for (int ns = 0; ns < 2; ++ns)
                acc[ms][ns] = __builtin_amdgcn_wmma_f32_16x16x32_bf16(
                    false, fa[ms].v, false, fb[ns].v,
                    (short)0, acc[ms][ns], false, false);

        __syncthreads();   // everyone done reading buf before it is refilled
    }

    const float s = *scale;
#pragma unroll
    for (int ms = 0; ms < 4; ++ms) {
#pragma unroll
        for (int ns = 0; ns < 2; ++ns) {
            int col = n0 + wn + ns * 16 + l16;
            float bv = bias[col];
#pragma unroll
            for (int r = 0; r < 8; ++r) {
                int row = m0 + wm + ms * 16 + half * 8 + r;   // C VGPR layout
                C[(size_t)row * N + col] = acc[ms][ns][r] * s + bv;
            }
        }
    }
}

// --------------------- LayerNorm + exact GELU -> bf16 -----------------------
__global__ __launch_bounds__(256)
void k_ln_gelu(const float* __restrict__ Y, const float* __restrict__ g,
               const float* __restrict__ be, unsigned short* __restrict__ H,
               int cols) {
    __shared__ float rs[256], rq[256];
    int row = blockIdx.x;
    int nc = cols >> 8;                 // 16 (H=4096) or 8 (H=2048)
    float v[16];
    float s = 0.f, q = 0.f;
    const float* yr = Y + (size_t)row * cols;
    for (int i = 0; i < nc; ++i) {
        float x = yr[threadIdx.x + (i << 8)];
        v[i] = x; s += x; q += x * x;
    }
    rs[threadIdx.x] = s; rq[threadIdx.x] = q;
    __syncthreads();
    for (int off = 128; off > 0; off >>= 1) {
        if ((int)threadIdx.x < off) {
            rs[threadIdx.x] += rs[threadIdx.x + off];
            rq[threadIdx.x] += rq[threadIdx.x + off];
        }
        __syncthreads();
    }
    float mu   = rs[0] / (float)cols;
    float var  = rq[0] / (float)cols - mu * mu;
    float rstd = rsqrtf(var + 1e-5f);
    unsigned short* hr = H + (size_t)row * cols;
    for (int i = 0; i < nc; ++i) {
        int c = threadIdx.x + (i << 8);
        float y  = (v[i] - mu) * rstd * g[c] + be[c];
        float ge = 0.5f * y * (1.0f + erff(y * 0.70710678118654752f));
        hr[c] = f2bf(ge);
    }
}

// ------------------------- classifier head (N=3) ----------------------------
__global__ __launch_bounds__(256)
void k_logits(const unsigned short* __restrict__ H, const unsigned short* __restrict__ W3,
              const float* __restrict__ scale, const float* __restrict__ b3,
              float* __restrict__ out) {
    int lane = threadIdx.x & 31;
    int row  = blockIdx.x * 8 + (threadIdx.x >> 5);
    const unsigned short* h = H + (size_t)row * HALFH;
    float p0 = 0.f, p1 = 0.f, p2 = 0.f;
    for (int k = lane; k < HALFH; k += 32) {
        float x = bf2f(h[k]);
        p0 += x * bf2f(W3[k]);
        p1 += x * bf2f(W3[HALFH + k]);
        p2 += x * bf2f(W3[2 * HALFH + k]);
    }
    for (int off = 16; off > 0; off >>= 1) {
        p0 += __shfl_xor(p0, off, 32);
        p1 += __shfl_xor(p1, off, 32);
        p2 += __shfl_xor(p2, off, 32);
    }
    if (lane == 0) {
        float s = *scale;
        out[(size_t)row * 3 + 0] = p0 * s + b3[0];
        out[(size_t)row * 3 + 1] = p1 * s + b3[1];
        out[(size_t)row * 3 + 2] = p2 * s + b3[2];
    }
}

// ---------------------------------------------------------------------------
extern "C" void kernel_launch(void* const* d_in, const int* in_sizes, int n_in,
                              void* d_out, int out_size, void* d_ws, size_t ws_size,
                              hipStream_t stream) {
    const int*   ids = (const int*)d_in[0];
    const float* emb = (const float*)d_in[1];
    const float* w1  = (const float*)d_in[2];
    const float* b1  = (const float*)d_in[3];
    const float* w2  = (const float*)d_in[4];
    const float* b2  = (const float*)d_in[5];
    const float* w3  = (const float*)d_in[6];
    const float* b3  = (const float*)d_in[7];
    const float* g1  = (const float*)d_in[8];
    const float* be1 = (const float*)d_in[9];
    const float* g2  = (const float*)d_in[10];
    const float* be2 = (const float*)d_in[11];
    float* out = (float*)d_out;
    char*  ws  = (char*)d_ws;

    const int n1 = HIDDEN * EMBED;   // 4,194,304
    const int n2 = HALFH * HIDDEN;   // 8,388,608
    const int n3 = NCLS * HALFH;     // 6,144

    // workspace layout
    float* scales = (float*)(ws);            // 3 f32
    float* P1     = (float*)(ws + 1024);     // 1024 partials
    float* P2     = (float*)(ws + 8192);     // 2048 partials
    float* P3     = (float*)(ws + 16384);    // 2 partials
    size_t o = 65536;
    unsigned short* W1Q = (unsigned short*)(ws + o); o += (size_t)n1 * 2;          // 8 MB
    unsigned short* W2Q = (unsigned short*)(ws + o); o += (size_t)n2 * 2;          // 16 MB
    unsigned short* W3Q = (unsigned short*)(ws + o); o += 16384;
    unsigned short* X   = (unsigned short*)(ws + o); o += (size_t)BATCH * EMBED * 2;   // 8 MB
    float*          Y   = (float*)(ws + o);          o += (size_t)BATCH * HIDDEN * 4;  // 64 MB (reused)
    unsigned short* H1  = (unsigned short*)(ws + o); o += (size_t)BATCH * HIDDEN * 2;  // 32 MB
    unsigned short* H2  = (unsigned short*)(ws + o);                                   // 16 MB

    // 1) per-tensor ternary scales (deterministic tree reductions)
    k_abs_partial<<<n1 / 4096, 256, 0, stream>>>(w1, n1, P1);
    k_abs_partial<<<n2 / 4096, 256, 0, stream>>>(w2, n2, P2);
    k_abs_partial<<<(n3 + 4095) / 4096, 256, 0, stream>>>(w3, n3, P3);
    k_finalize_scales<<<1, 256, 0, stream>>>(P1, n1 / 4096, 1.f / n1,
                                             P2, n2 / 4096, 1.f / n2,
                                             P3, (n3 + 4095) / 4096, 1.f / n3, scales);
    // 2) quantize weights to bf16 {-1,0,+1}
    k_ternary<<<(n1 + 1023) / 1024, 256, 0, stream>>>(w1, n1, scales + 0, W1Q);
    k_ternary<<<(n2 + 1023) / 1024, 256, 0, stream>>>(w2, n2, scales + 1, W2Q);
    k_ternary<<<(n3 + 1023) / 1024, 256, 0, stream>>>(w3, n3, scales + 2, W3Q);
    // 3) embedding gather + masked mean pool (bandwidth-bound stage)
    k_pool<<<BATCH, 256, 0, stream>>>(ids, emb, X);
    // 4) layer 1: WMMA GEMM + LN + GELU
    dim3 gg1(BATCH / 128, HIDDEN / 128);
    k_gemm_bf16<<<gg1, 256, 0, stream>>>(X, W1Q, Y, BATCH, HIDDEN, EMBED, scales + 0, b1);
    k_ln_gelu<<<BATCH, 256, 0, stream>>>(Y, g1, be1, H1, HIDDEN);
    // 5) layer 2
    dim3 gg2(BATCH / 128, HALFH / 128);
    k_gemm_bf16<<<gg2, 256, 0, stream>>>(H1, W2Q, Y, BATCH, HALFH, HIDDEN, scales + 1, b2);
    k_ln_gelu<<<BATCH, 256, 0, stream>>>(Y, g2, be2, H2, HALFH);
    // 6) classifier head
    k_logits<<<BATCH / 8, 256, 0, stream>>>(H2, W3Q, scales + 2, b3, out);
}
